// SpikeNet_45887430590690
// MI455X (gfx1250) — compile-verified, matrix operand order checked by reading, not run on hardware
//
#include <hip/hip_runtime.h>

// ---------------- problem constants (from reference) ----------------
#define TT   16      // timesteps
#define NN   50000   // nodes
#define FF   128     // features
#define BB   1024    // batch
#define S0C  10      // 1-hop fanout
#define S1C  5       // 2-hop fanout
#define H0C  128
#define H1C  64
#define CC   10
#define KC   5
#define RR   (BB + BB * S0C)   // 11264 rows in layer-0
#define VTHF 1.0f

typedef _Float16 h16;
typedef __attribute__((ext_vector_type(4)))  _Float16 v4h;
typedef __attribute__((ext_vector_type(8)))  _Float16 v8h;
typedef __attribute__((ext_vector_type(16))) _Float16 v16h;
typedef __attribute__((ext_vector_type(8)))  float    v8f;

// ---------------- fragment load helper ----------------
// 16-bit A/B fragment per CDNA5 layout:
//   lanes 0-15 : row r, K = {k0..k0+7, k0+16..k0+23}
//   lanes 16-31: row r, K = {k0+8..k0+15, k0+24..k0+31}
// caller passes p = row_base + k0 + (lane>>4)*8
__device__ __forceinline__ v16h load_frag(const h16* __restrict__ p) {
    v8h lo = *(const v8h*)(p);
    v8h hi = *(const v8h*)(p + 16);
    v16h f;
#pragma unroll
    for (int i = 0; i < 8; ++i) { f[i] = lo[i]; f[i + 8] = hi[i]; }
    return f;
}

// ---------------- kernel 1: weight convert + transpose to f16 ----------------
// WtT[n][k] = W[k][n]; makes B-fragment loads contiguous per lane.
__global__ __launch_bounds__(256) void prep_weights_kernel(
    const float* __restrict__ Ws0, const float* __restrict__ Wn0,
    const float* __restrict__ Ws1, const float* __restrict__ Wn1,
    h16* __restrict__ Ws0T, h16* __restrict__ Wn0T,
    h16* __restrict__ Ws1T, h16* __restrict__ Wn1T)
{
    int id = blockIdx.x * blockDim.x + threadIdx.x;
    if (id < FF * H0C) {                 // 128x128 -> 128x128
        int n = id / FF, k = id % FF;
        Ws0T[id] = (h16)Ws0[k * H0C + n];
        Wn0T[id] = (h16)Wn0[k * H0C + n];
    }
    if (id < H1C * H0C) {                // 128x64 -> 64x128
        int n = id / H0C, k = id % H0C;
        Ws1T[id] = (h16)Ws1[k * H1C + n];
        Wn1T[id] = (h16)Wn1[k * H1C + n];
    }
}

// ---------------- kernel 2: fold delays+depthwise conv+projection into M ----
// M[s][ch][c]; out[b,c] = b_c + sum_{s,ch} spk1[s,b,ch] * M[s,ch,c]
__global__ void coef_kernel(const float* __restrict__ delay_logits,
                            const float* __restrict__ dw,
                            const float* __restrict__ pw,
                            float* __restrict__ M)
{
    int ch = threadIdx.x;
    if (ch >= H1C) return;
    int g = ch >> 3;                          // H1/G = 8 channels per group
    float l0 = delay_logits[g * 3 + 0];
    float l1 = delay_logits[g * 3 + 1];
    float l2 = delay_logits[g * 3 + 2];
    float mx = fmaxf(l0, fmaxf(l1, l2));
    float e0 = __expf(l0 - mx), e1 = __expf(l1 - mx), e2 = __expf(l2 - mx);
    float inv = 1.0f / (e0 + e1 + e2);
    float w[3] = { e0 * inv, e1 * inv, e2 * inv };
    const int D[3] = { 1, 3, 5 };
    for (int s = 0; s < TT; ++s) {
        float coef = 0.0f;
        for (int i = 0; i < 3; ++i) {
            int u = s + D[i];
            if (u >= TT) continue;            // delayed index in range
            for (int k = 0; k < KC; ++k) {
                int t = u - k + 2;            // SAME conv output index
                if (t < 0 || t >= TT) continue;
                coef += w[i] * dw[ch * KC + k];
            }
        }
        float sc = coef * (1.0f / (float)TT);
        for (int c = 0; c < CC; ++c)
            M[(s * H1C + ch) * CC + c] = sc * pw[ch * CC + c];
    }
}

// ---------------- kernel 3: gather + neighbor means -> f16 (per t) ----------
// one wave per output row; lane holds 4 contiguous columns (float4 loads)
__global__ __launch_bounds__(256) void gather_kernel(
    const float* __restrict__ x, const int* __restrict__ nodes,
    const int* __restrict__ nbr1, const int* __restrict__ nbr2,
    h16* __restrict__ selfx, h16* __restrict__ neigh, int t)
{
    int wid  = (blockIdx.x * blockDim.x + threadIdx.x) >> 5;
    int lane = threadIdx.x & 31;
    const float* xt = x + (size_t)t * NN * FF;
    int col = lane * 4;
    float a0 = 0.f, a1 = 0.f, a2 = 0.f, a3 = 0.f;
    h16* dst;

    if (wid < BB) {                                       // self rows of batch
        const float* p = xt + (size_t)nodes[wid] * FF + col;
        float4 v = *(const float4*)p;
        a0 = v.x; a1 = v.y; a2 = v.z; a3 = v.w;
        dst = selfx + (size_t)wid * FF;
    } else if (wid < RR) {                                // self rows of 1-hop
        int i = wid - BB;
        const float* p = xt + (size_t)nbr1[t * BB * S0C + i] * FF + col;
        float4 v = *(const float4*)p;
        a0 = v.x; a1 = v.y; a2 = v.z; a3 = v.w;
        dst = selfx + (size_t)wid * FF;
    } else if (wid < RR + BB) {                           // neigh mean over S0
        int b = wid - RR;
        for (int s = 0; s < S0C; ++s) {
            const float* p = xt + (size_t)nbr1[t * BB * S0C + b * S0C + s] * FF + col;
            float4 v = *(const float4*)p;
            a0 += v.x; a1 += v.y; a2 += v.z; a3 += v.w;
        }
        a0 *= 0.1f; a1 *= 0.1f; a2 *= 0.1f; a3 *= 0.1f;
        dst = neigh + (size_t)b * FF;
    } else {                                              // neigh mean over S1
        int i = wid - RR - BB;
        for (int s = 0; s < S1C; ++s) {
            const float* p = xt + (size_t)nbr2[t * BB * S0C * S1C + i * S1C + s] * FF + col;
            float4 v = *(const float4*)p;
            a0 += v.x; a1 += v.y; a2 += v.z; a3 += v.w;
        }
        a0 *= 0.2f; a1 *= 0.2f; a2 *= 0.2f; a3 *= 0.2f;
        dst = neigh + (size_t)(BB + i) * FF;
    }
    v4h o; o[0] = (h16)a0; o[1] = (h16)a1; o[2] = (h16)a2; o[3] = (h16)a3;
    *(v4h*)(dst + col) = o;
}

// ---------------- kernel 4: layer-0 GEMM + spike (WMMA f16->f32) -----------
// a0 = selfx@Ws0 + neigh@Wn0 + b0 ; spk0 = (a0 >= VTH)
// 16x16 tile per wave; 704 m-tiles x 8 n-tiles = 5632 waves = 704 blocks
__global__ __launch_bounds__(256) void gemm0_kernel(
    const h16* __restrict__ selfx, const h16* __restrict__ neigh,
    const h16* __restrict__ Ws0T,  const h16* __restrict__ Wn0T,
    const float* __restrict__ b0,  h16* __restrict__ spk0)
{
    int wave = (blockIdx.x * blockDim.x + threadIdx.x) >> 5;
    int lane = threadIdx.x & 31;
    int m0 = (wave >> 3) * 16;
    int n0 = (wave & 7) * 16;
    int half8 = (lane >> 4) * 8;
    int am = m0 + (lane & 15);
    int bn = n0 + (lane & 15);

    v8f acc = {};
#pragma unroll
    for (int kc = 0; kc < 4; ++kc) {
        int k0 = kc * 32;
        v16h a = load_frag(selfx + (size_t)am * FF + k0 + half8);
        v16h b = load_frag(Ws0T  + (size_t)bn * FF + k0 + half8);
        acc = __builtin_amdgcn_wmma_f32_16x16x32_f16(false, a, false, b,
                                                     (short)0, acc, false, false);
    }
#pragma unroll
    for (int kc = 0; kc < 4; ++kc) {
        int k0 = kc * 32;
        v16h a = load_frag(neigh + (size_t)am * FF + k0 + half8);
        v16h b = load_frag(Wn0T  + (size_t)bn * FF + k0 + half8);
        acc = __builtin_amdgcn_wmma_f32_16x16x32_f16(false, a, false, b,
                                                     (short)0, acc, false, false);
    }
    // C/D layout: VGPR j -> row m0 + j + (lane>=16 ? 8 : 0), col n0 + (lane&15)
    int col  = n0 + (lane & 15);
    int row0 = m0 + ((lane >> 4) << 3);
    float bias = b0[col];
#pragma unroll
    for (int j = 0; j < 8; ++j) {
        float v = acc[j] + bias;
        spk0[(size_t)(row0 + j) * H0C + col] = (v >= VTHF) ? (h16)1.0f : (h16)0.0f;
    }
}

// ---------------- kernel 5: mean of 1-hop spikes over S0 -------------------
__global__ __launch_bounds__(256) void smean_kernel(
    const h16* __restrict__ spk0, h16* __restrict__ smean)
{
    int wid  = (blockIdx.x * blockDim.x + threadIdx.x) >> 5;   // row b
    int lane = threadIdx.x & 31;
    int col = lane * 4;
    float a0 = 0.f, a1 = 0.f, a2 = 0.f, a3 = 0.f;
    for (int s = 0; s < S0C; ++s) {
        v4h v = *(const v4h*)(spk0 + (size_t)(BB + wid * S0C + s) * H0C + col);
        a0 += (float)v[0]; a1 += (float)v[1]; a2 += (float)v[2]; a3 += (float)v[3];
    }
    v4h o;
    o[0] = (h16)(a0 * 0.1f); o[1] = (h16)(a1 * 0.1f);
    o[2] = (h16)(a2 * 0.1f); o[3] = (h16)(a3 * 0.1f);
    *(v4h*)(smean + (size_t)wid * H0C + col) = o;
}

// ---------------- kernel 6: layer-1 GEMM + spike (WMMA) --------------------
// a1 = s_self@Ws1 + smean@Wn1 + b1 ; spk1 -> spikes[t] (f32 0/1)
// 64 m-tiles x 4 n-tiles = 256 waves = 32 blocks
__global__ __launch_bounds__(256) void gemm1_kernel(
    const h16* __restrict__ sself, const h16* __restrict__ smean,
    const h16* __restrict__ Ws1T,  const h16* __restrict__ Wn1T,
    const float* __restrict__ b1,  float* __restrict__ spikes_t)
{
    int wave = (blockIdx.x * blockDim.x + threadIdx.x) >> 5;
    int lane = threadIdx.x & 31;
    int m0 = (wave >> 2) * 16;
    int n0 = (wave & 3) * 16;
    int half8 = (lane >> 4) * 8;
    int am = m0 + (lane & 15);
    int bn = n0 + (lane & 15);

    v8f acc = {};
#pragma unroll
    for (int kc = 0; kc < 4; ++kc) {
        int k0 = kc * 32;
        v16h a = load_frag(sself + (size_t)am * H0C + k0 + half8);
        v16h b = load_frag(Ws1T  + (size_t)bn * H0C + k0 + half8);
        acc = __builtin_amdgcn_wmma_f32_16x16x32_f16(false, a, false, b,
                                                     (short)0, acc, false, false);
    }
#pragma unroll
    for (int kc = 0; kc < 4; ++kc) {
        int k0 = kc * 32;
        v16h a = load_frag(smean + (size_t)am * H0C + k0 + half8);
        v16h b = load_frag(Wn1T  + (size_t)bn * H0C + k0 + half8);
        acc = __builtin_amdgcn_wmma_f32_16x16x32_f16(false, a, false, b,
                                                     (short)0, acc, false, false);
    }
    int col  = n0 + (lane & 15);
    int row0 = m0 + ((lane >> 4) << 3);
    float bias = b1[col];
#pragma unroll
    for (int j = 0; j < 8; ++j) {
        float v = acc[j] + bias;
        spikes_t[(size_t)(row0 + j) * H1C + col] = (v >= VTHF) ? 1.0f : 0.0f;
    }
}

// ---------------- kernel 7: folded temporal readout ------------------------
__global__ void readout_kernel(const float* __restrict__ spikes,
                               const float* __restrict__ M,
                               const float* __restrict__ tsr_b,
                               float* __restrict__ out)
{
    int b = blockIdx.x;
    int c = threadIdx.x;
    if (c >= CC) return;
    float acc = tsr_b[c];
    for (int t = 0; t < TT; ++t) {
        const float* sp = spikes + ((size_t)t * BB + b) * H1C;
        const float* mm = M + (size_t)t * H1C * CC + c;
        for (int ch = 0; ch < H1C; ++ch)
            acc += sp[ch] * mm[ch * CC];
    }
    out[b * CC + c] = acc;
}

// ---------------- host-side orchestration ----------------------------------
extern "C" void kernel_launch(void* const* d_in, const int* in_sizes, int n_in,
                              void* d_out, int out_size, void* d_ws, size_t ws_size,
                              hipStream_t stream) {
    (void)in_sizes; (void)n_in; (void)out_size; (void)ws_size;

    const float* x    = (const float*)d_in[0];
    const int*   nodes= (const int*)  d_in[1];
    const int*   nbr1 = (const int*)  d_in[2];
    const int*   nbr2 = (const int*)  d_in[3];
    const float* Ws0  = (const float*)d_in[4];
    const float* Wn0  = (const float*)d_in[5];
    const float* b0   = (const float*)d_in[6];
    const float* Ws1  = (const float*)d_in[7];
    const float* Wn1  = (const float*)d_in[8];
    const float* b1   = (const float*)d_in[9];
    const float* dlog = (const float*)d_in[10];
    const float* dw   = (const float*)d_in[11];
    const float* pw   = (const float*)d_in[12];
    const float* tb   = (const float*)d_in[13];
    float* out = (float*)d_out;

    // workspace carve-up (256B aligned)
    char* ws = (char*)d_ws;
    size_t off = 0;
    auto carve = [&](size_t bytes) { char* p = ws + off; off = (off + bytes + 255) & ~(size_t)255; return p; };
    h16*   Ws0T   = (h16*)  carve((size_t)H0C * FF  * 2);      // 128x128
    h16*   Wn0T   = (h16*)  carve((size_t)H0C * FF  * 2);
    h16*   Ws1T   = (h16*)  carve((size_t)H1C * H0C * 2);      // 64x128
    h16*   Wn1T   = (h16*)  carve((size_t)H1C * H0C * 2);
    h16*   selfx  = (h16*)  carve((size_t)RR  * FF  * 2);
    h16*   neigh  = (h16*)  carve((size_t)RR  * FF  * 2);
    h16*   spk0   = (h16*)  carve((size_t)RR  * H0C * 2);
    h16*   smean  = (h16*)  carve((size_t)BB  * H0C * 2);
    float* spikes = (float*)carve((size_t)TT * BB * H1C * 4);
    float* Mcoef  = (float*)carve((size_t)TT * H1C * CC * 4);

    // one-time prep
    prep_weights_kernel<<<(FF * H0C + 255) / 256, 256, 0, stream>>>(
        Ws0, Wn0, Ws1, Wn1, Ws0T, Wn0T, Ws1T, Wn1T);
    coef_kernel<<<1, 64, 0, stream>>>(dlog, dw, pw, Mcoef);

    // TAU==1 makes steps independent; loop t to keep workspace small
    const int gatherWaves = RR + BB + BB * S0C;          // 22528
    for (int t = 0; t < TT; ++t) {
        gather_kernel<<<(gatherWaves * 32 + 255) / 256, 256, 0, stream>>>(
            x, nodes, nbr1, nbr2, selfx, neigh, t);
        gemm0_kernel<<<(RR / 16) * 8 / 8, 256, 0, stream>>>(   // 704 blocks
            selfx, neigh, Ws0T, Wn0T, b0, spk0);
        smean_kernel<<<BB * 32 / 256, 256, 0, stream>>>(spk0, smean);
        gemm1_kernel<<<(BB / 16) * (H1C / 16) / 8, 256, 0, stream>>>( // 32 blocks
            spk0 /* rows 0..B-1 are s_self */, smean, Ws1T, Wn1T, b1,
            spikes + (size_t)t * BB * H1C);
    }

    readout_kernel<<<BB, 32, 0, stream>>>(spikes, Mcoef, tb, out);
}
